// DifferentiableMultiMLPRenderer_3951369912801
// MI455X (gfx1250) — matrix operand order repaired and checked by reading, not compiled, not run on hardware
//
#include <hip/hip_runtime.h>
#include <cstdint>
#include <cstddef>

// ---------------- types ----------------
typedef __attribute__((ext_vector_type(16))) __bf16 v16bf;
typedef __attribute__((ext_vector_type(8)))  float  v8f;

union FragB { uint4 u[2]; v16bf v; };

__device__ __forceinline__ unsigned short f2bf(float f) {
    union { float f; unsigned u; } x; x.f = f;
    unsigned u = x.u;
    unsigned r = (u + 0x7FFFu + ((u >> 16) & 1u)) >> 16;   // round-to-nearest-even
    return (unsigned short)r;
}
__device__ __forceinline__ float bf2f(unsigned short s) {
    union { unsigned u; float f; } x; x.u = ((unsigned)s) << 16; return x.f;
}

// ---------------- constants ----------------
#define D_IN   703
#define K1     704          // padded K for layer 1
#define HID    256
#define XSTR   720          // LDS row stride (bf16) for input tile
#define HSTR   264          // LDS row stride (bf16) for hidden tiles
#define MPIX   64           // pixels per workgroup (4 M-tiles of 16)
#define MT     4            // M-tiles per wave

// ---------------- weight conversion: f32 [K][N] -> bf16 transposed [N][K] ----------------
__global__ __launch_bounds__(256)
void convert_weights(const float* __restrict__ w1, const float* __restrict__ w2,
                     unsigned short* __restrict__ W1b, unsigned short* __restrict__ W2b)
{
    int i = blockIdx.x * 256 + threadIdx.x;
    const int N1 = HID * K1;          // 180224
    const int N2 = HID * HID;         // 65536
    if (i < N1) {
        int n = i / K1, k = i % K1;
        W1b[i] = (k < D_IN) ? f2bf(w1[k * HID + n]) : (unsigned short)0;
    } else if (i < N1 + N2) {
        int j = i - N1;
        int n = j / HID, k = j % HID;
        W2b[j] = f2bf(w2[k * HID + n]);
    }
}

// ---------------- main renderer ----------------
__global__ __launch_bounds__(256)
void renderer_kernel(const int*   __restrict__ pix_to_face,
                     const float* __restrict__ bary,
                     const int*   __restrict__ faces,
                     const float* __restrict__ feature,
                     const float* __restrict__ shapef,
                     const float* __restrict__ color_bg,
                     const float* __restrict__ style,
                     const float* __restrict__ b1,
                     const float* __restrict__ b2,
                     const float* __restrict__ w3,
                     const float* __restrict__ b3,
                     const unsigned short* __restrict__ W1b,
                     const unsigned short* __restrict__ W2b,
                     float* __restrict__ out,
                     int hw)
{
    __shared__ __align__(16) unsigned short sX[MPIX * XSTR];   // 64x704(+pad) bf16 input tile
    __shared__ __align__(16) unsigned short sH1[MPIX * HSTR];  // 64x256(+pad) bf16 hidden 1
    __shared__ __align__(16) unsigned short sH2[MPIX * HSTR];  // 64x256(+pad) bf16 hidden 2
    __shared__ float sC[MPIX * 3];

    const int tid  = threadIdx.x;
    const int pix0 = blockIdx.x * MPIX;

    // ---------- phase 1: build bf16 input tile in LDS (4 threads per pixel) ----------
    {
        const int p = tid >> 2;          // local pixel 0..63
        const int l = tid & 3;           // 4 threads cooperate per pixel
        const int g = pix0 + p;
        const int face = pix_to_face[g];
        const int v0 = faces[face * 3 + 0];
        const int v1 = faces[face * 3 + 1];
        const int v2 = faces[face * 3 + 2];
        const float c0 = bary[g * 3 + 0];
        const float c1 = bary[g * 3 + 1];
        const float c2 = bary[g * 3 + 2];

        // vertex feature (256) -> cols [0,256)
        for (int c = l; c < 256; c += 4) {
            float v = (c0 * feature[v0 * 256 + c] +
                       c1 * feature[v1 * 256 + c] +
                       c2 * feature[v2 * 256 + c]) * (1.0f / 3.0f);
            sX[p * XSTR + c] = f2bf(v);
        }
        // shape feature (128) -> cols [256,384)
        for (int c = l; c < 128; c += 4) {
            float v = (c0 * shapef[v0 * 128 + c] +
                       c1 * shapef[v1 * 128 + c] +
                       c2 * shapef[v2 * 128 + c]) * (1.0f / 3.0f);
            sX[p * XSTR + 256 + c] = f2bf(v);
        }
        // NeRF embed (63) -> cols [384,447): [x(3), sin(f0 x)(3), cos(f0 x)(3), ...]
        for (int e = l; e < 63; e += 4) {
            float v;
            if (e < 3) {
                v = (e == 0) ? c0 : ((e == 1) ? c1 : c2);
            } else {
                int t = e - 3;
                int fr = t / 6;          // frequency index
                int r  = t % 6;
                int s  = r / 3;          // 0 = sin, 1 = cos
                int d  = r % 3;          // coord
                float x = ((d == 0) ? c0 : ((d == 1) ? c1 : c2)) * (float)(1u << fr);
                v = (s == 0) ? __sinf(x) : __cosf(x);
            }
            sX[p * XSTR + 384 + e] = f2bf(v);
        }
        // style latent (256) -> cols [447,703)
        for (int c = l; c < 256; c += 4)
            sX[p * XSTR + 447 + c] = f2bf(style[c]);
        // pad col 703
        if (l == 0) sX[p * XSTR + D_IN] = 0;
    }
    __syncthreads();

    const int lane = tid & 31;
    const int wave = tid >> 5;           // 8 waves; wave owns output cols [32w, 32w+32)
    const int lrow = lane & 15;          // A row / B col / D col index
    const int half = lane >> 4;

    // ---------- phase 2: layer 1 -- H1[64,256] = relu(X @ W1 + b1), K = 704 ----------
    {
        v8f acc0[MT] = {{}, {}, {}, {}};     // N-tile 0, M-tiles 0..3
        v8f acc1[MT] = {{}, {}, {}, {}};     // N-tile 1, M-tiles 0..3
        for (int ks = 0; ks < K1; ks += 32) {
            FragB Ba, Bb;
            const unsigned short* bp0 = &W1b[(wave * 32 + lrow) * K1 + ks + 16 * half];
            const unsigned short* bp1 = &W1b[(wave * 32 + 16 + lrow) * K1 + ks + 16 * half];
            Ba.u[0] = *(const uint4*)(bp0);
            Ba.u[1] = *(const uint4*)(bp0 + 8);
            Bb.u[0] = *(const uint4*)(bp1);
            Bb.u[1] = *(const uint4*)(bp1 + 8);

            #pragma unroll
            for (int t = 0; t < MT; ++t) {
                // A layout: lanes 0-15 -> K {ks..ks+15}; lanes 16-31 -> +16
                FragB A;
                const int row = t * 16 + lrow;
                A.u[0] = *(const uint4*)(&sX[row * XSTR + ks + 16 * half]);
                A.u[1] = *(const uint4*)(&sX[row * XSTR + ks + 16 * half + 8]);
                acc0[t] = __builtin_amdgcn_wmma_f32_16x16x32_bf16(false, A.v, false, Ba.v,
                                                                  (short)0, acc0[t], false, false);
                acc1[t] = __builtin_amdgcn_wmma_f32_16x16x32_bf16(false, A.v, false, Bb.v,
                                                                  (short)0, acc1[t], false, false);
            }
        }
        // epilogue: bias + relu -> bf16 LDS
        const int col0 = wave * 32 + lrow;
        const int col1 = wave * 32 + 16 + lrow;
        const float bia0 = b1[col0], bia1 = b1[col1];
        #pragma unroll
        for (int t = 0; t < MT; ++t) {
            #pragma unroll
            for (int r = 0; r < 8; ++r) {
                const int m = t * 16 + r + 8 * half;
                float v;
                v = acc0[t][r] + bia0; v = v > 0.f ? v : 0.f; sH1[m * HSTR + col0] = f2bf(v);
                v = acc1[t][r] + bia1; v = v > 0.f ? v : 0.f; sH1[m * HSTR + col1] = f2bf(v);
            }
        }
    }
    __syncthreads();

    // ---------- phase 3: layer 2 -- H2[64,256] = relu(H1 @ W2 + b2), K = 256 ----------
    {
        v8f acc0[MT] = {{}, {}, {}, {}};
        v8f acc1[MT] = {{}, {}, {}, {}};
        for (int ks = 0; ks < HID; ks += 32) {
            FragB Ba, Bb;
            const unsigned short* bp0 = &W2b[(wave * 32 + lrow) * HID + ks + 16 * half];
            const unsigned short* bp1 = &W2b[(wave * 32 + 16 + lrow) * HID + ks + 16 * half];
            Ba.u[0] = *(const uint4*)(bp0);
            Ba.u[1] = *(const uint4*)(bp0 + 8);
            Bb.u[0] = *(const uint4*)(bp1);
            Bb.u[1] = *(const uint4*)(bp1 + 8);

            #pragma unroll
            for (int t = 0; t < MT; ++t) {
                FragB A;
                const int row = t * 16 + lrow;
                A.u[0] = *(const uint4*)(&sH1[row * HSTR + ks + 16 * half]);
                A.u[1] = *(const uint4*)(&sH1[row * HSTR + ks + 16 * half + 8]);
                acc0[t] = __builtin_amdgcn_wmma_f32_16x16x32_bf16(false, A.v, false, Ba.v,
                                                                  (short)0, acc0[t], false, false);
                acc1[t] = __builtin_amdgcn_wmma_f32_16x16x32_bf16(false, A.v, false, Bb.v,
                                                                  (short)0, acc1[t], false, false);
            }
        }
        const int col0 = wave * 32 + lrow;
        const int col1 = wave * 32 + 16 + lrow;
        const float bia0 = b2[col0], bia1 = b2[col1];
        #pragma unroll
        for (int t = 0; t < MT; ++t) {
            #pragma unroll
            for (int r = 0; r < 8; ++r) {
                const int m = t * 16 + r + 8 * half;
                float v;
                v = acc0[t][r] + bia0; v = v > 0.f ? v : 0.f; sH2[m * HSTR + col0] = f2bf(v);
                v = acc1[t][r] + bia1; v = v > 0.f ? v : 0.f; sH2[m * HSTR + col1] = f2bf(v);
            }
        }
    }
    __syncthreads();

    // ---------- phase 4: layer 3 (256 -> 3), relu, -1 ----------
    if (tid < MPIX * 3) {
        const int p  = tid / 3;
        const int ch = tid % 3;
        float acc = b3[ch];
        for (int k = 0; k < HID; ++k)
            acc += bf2f(sH2[p * HSTR + k]) * w3[k * 3 + ch];
        acc = (acc > 0.f ? acc : 0.f) - 1.0f;
        sC[p * 3 + ch] = acc;
    }
    __syncthreads();

    // ---------- phase 5: composite + write [g,4] ----------
    if (tid < MPIX) {
        const int g = pix0 + tid;
        const bool m = pix_to_face[g] > 0;
        const int b = g / hw;
        float o0, o1, o2;
        if (m) {
            o0 = sC[tid * 3 + 0]; o1 = sC[tid * 3 + 1]; o2 = sC[tid * 3 + 2];
        } else {
            o0 = color_bg[b * 3 + 0]; o1 = color_bg[b * 3 + 1]; o2 = color_bg[b * 3 + 2];
        }
        float4 o = make_float4(o0, o1, o2, m ? 1.0f : 0.0f);
        *(float4*)(&out[(size_t)g * 4]) = o;
    }
}

// ---------------- launch ----------------
extern "C" void kernel_launch(void* const* d_in, const int* in_sizes, int n_in,
                              void* d_out, int out_size, void* d_ws, size_t ws_size,
                              hipStream_t stream)
{
    const int*   pix_to_face = (const int*)  d_in[0];
    const float* bary        = (const float*)d_in[1];
    const int*   faces       = (const int*)  d_in[2];
    const float* feature     = (const float*)d_in[3];
    const float* shapef      = (const float*)d_in[4];
    const float* color_bg    = (const float*)d_in[5];
    const float* style       = (const float*)d_in[6];
    const float* w1          = (const float*)d_in[7];
    const float* b1          = (const float*)d_in[8];
    const float* w2          = (const float*)d_in[9];
    const float* b2          = (const float*)d_in[10];
    const float* w3          = (const float*)d_in[11];
    const float* b3          = (const float*)d_in[12];

    unsigned short* W1b = (unsigned short*)d_ws;                          // 256*704 bf16
    unsigned short* W2b = (unsigned short*)((char*)d_ws + HID * K1 * 2);  // 256*256 bf16 (16B aligned)

    const int npix = in_sizes[0];                 // B*H*W
    const int Bv   = in_sizes[5] / 3;             // color_bg has B*3 elements
    const int hw   = npix / Bv;

    const int conv_elems  = HID * K1 + HID * HID; // 245760
    const int conv_blocks = (conv_elems + 255) / 256;
    convert_weights<<<conv_blocks, 256, 0, stream>>>(w1, w2, W1b, W2b);

    renderer_kernel<<<npix / MPIX, 256, 0, stream>>>(
        pix_to_face, bary, faces, feature, shapef, color_bg, style,
        b1, b2, w3, b3, W1b, W2b, (float*)d_out, hw);
}
